// RelationAttnTemporalHTGNN_57647051047112
// MI455X (gfx1250) — matrix operand order repaired and checked by reading, not compiled, not run on hardware
//
#include <hip/hip_runtime.h>
#include <hip/hip_bf16.h>

// ---------------------------------------------------------------------------
// Problem constants (from the reference)
// ---------------------------------------------------------------------------
#define NN   50000
#define TT   4
#define RR   3
#define EE   500000
#define HID  128

typedef __attribute__((ext_vector_type(16))) _Float16 v16h;
typedef __attribute__((ext_vector_type(8)))  _Float16 v8h;
typedef __attribute__((ext_vector_type(8)))  float    v8f;

// ---------------------------------------------------------------------------
// Generic zero / convert kernels
// ---------------------------------------------------------------------------
__global__ void zero_f32(float* __restrict__ p, size_t n) {
    size_t i = (size_t)blockIdx.x * blockDim.x + threadIdx.x;
    if (i < n) p[i] = 0.0f;
}

__global__ void f32_to_f16(const float* __restrict__ in, _Float16* __restrict__ out, size_t n) {
    size_t i = (size_t)blockIdx.x * blockDim.x + threadIdx.x;
    if (i < n) out[i] = (_Float16)in[i];
}

// ---------------------------------------------------------------------------
// Pack a row-major f32 weight W[K][Kout] into the WMMA B-fragment layout:
//   out[((nt*KT + kt)*32 + lane)*16 + e]  with
//   k = kt*32 + (lane>>4)*8 + (e&7) + (e>>3)*16 ,  n = nt*16 + (lane&15)
// ---------------------------------------------------------------------------
__global__ void pack_b(const float* __restrict__ W, _Float16* __restrict__ out,
                       int K, int Kout, int total) {
    int idx = blockIdx.x * blockDim.x + threadIdx.x;
    if (idx >= total) return;
    int e    = idx & 15;
    int lane = (idx >> 4) & 31;
    int tile = idx >> 9;
    int KT   = K >> 5;
    int nt   = tile / KT;
    int kt   = tile - nt * KT;
    int k    = kt * 32 + (lane >> 4) * 8 + (e & 7) + (e >> 3) * 16;
    int n    = nt * 16 + (lane & 15);
    out[idx] = (_Float16)W[(size_t)k * Kout + n];
}

// ---------------------------------------------------------------------------
// WMMA GEMM:  C[N, Kout] = act(A[N, K](f16, lda) @ Bpacked + bias)
// Block = 256 threads = 8 waves; wave w owns output tile (blockIdx.x,
// blockIdx.y*8 + w). K-loop of v_wmma_f32_16x16x32_f16 with f32 accumulate.
// TWO independent accumulator chains (even/odd K-tiles) so consecutive
// v_wmma ops are not RAW-dependent -> compiler can fill the 4-slot WMMA
// hazard window with the other chain's WMMA instead of v_nops.
// K is always a multiple of 64 here (ktiles = 4 or 12).
// ---------------------------------------------------------------------------
__global__ void __launch_bounds__(256)
wmma_gemm(const _Float16* __restrict__ A, int lda, int ktiles,
          const _Float16* __restrict__ Bpk,
          const float* __restrict__ bias,
          float* __restrict__ Cf, _Float16* __restrict__ Ch,
          int ldc, int relu)
{
    int wave = threadIdx.x >> 5;
    int lane = threadIdx.x & 31;
    int mt   = blockIdx.x;
    int nt   = blockIdx.y * 8 + wave;
    int grp  = lane >> 4;
    int lrow = lane & 15;
    int row0 = mt * 16;
    int col0 = nt * 16;

    const _Float16* Arow = A + (size_t)(row0 + lrow) * lda;
    const _Float16* Bt   = Bpk + (size_t)nt * ktiles * 512 + lane * 16;

    v8f c0 = {};
    v8f c1 = {};
    for (int kt = 0; kt < ktiles; kt += 2) {
        // even K-tile -> chain 0
        int kb0 = kt * 32 + grp * 8;
        v8h alo0 = *(const v8h*)(Arow + kb0);
        v8h ahi0 = *(const v8h*)(Arow + kb0 + 16);
        v16h a0 = __builtin_shufflevector(alo0, ahi0,
                                          0, 1, 2, 3, 4, 5, 6, 7,
                                          8, 9, 10, 11, 12, 13, 14, 15);
        v16h b0 = *(const v16h*)(Bt + (size_t)kt * 512);
        // odd K-tile -> chain 1
        int kb1 = kb0 + 32;
        v8h alo1 = *(const v8h*)(Arow + kb1);
        v8h ahi1 = *(const v8h*)(Arow + kb1 + 16);
        v16h a1 = __builtin_shufflevector(alo1, ahi1,
                                          0, 1, 2, 3, 4, 5, 6, 7,
                                          8, 9, 10, 11, 12, 13, 14, 15);
        v16h b1 = *(const v16h*)(Bt + (size_t)(kt + 1) * 512);

        c0 = __builtin_amdgcn_wmma_f32_16x16x32_f16(
                false, a0, false, b0, (short)0, c0, false, false);
        c1 = __builtin_amdgcn_wmma_f32_16x16x32_f16(
                false, a1, false, b1, (short)0, c1, false, false);
    }
    v8f c = c0 + c1;

    // C layout: lane -> col = col0 + (lane&15); vgpr v -> row = row0 + grp*8 + v
    float bv = bias[col0 + lrow];
#pragma unroll
    for (int v = 0; v < 8; ++v) {
        int r = row0 + grp * 8 + v;
        float x = c[v] + bv;
        if (relu) x = fmaxf(x, 0.0f);
        size_t off = (size_t)r * ldc + col0 + lrow;
        if (Cf) Cf[off] = x;
        if (Ch) Ch[off] = (_Float16)x;
    }
}

// ---------------------------------------------------------------------------
// Edge scatter: one wave per edge, lane holds one float4 chunk of the row.
//   agg[r][dst] += ew * h[src];  deg[r][dst] += 1
// grid = ((E+7)/8, R)
// ---------------------------------------------------------------------------
__global__ void __launch_bounds__(256)
edge_scatter(const float* __restrict__ h,
             const int* __restrict__ src, const int* __restrict__ dst,
             const float* __restrict__ ew,
             float* __restrict__ agg, float* __restrict__ deg, int E)
{
    int r = blockIdx.y;
    int e = blockIdx.x * 8 + (threadIdx.x >> 5);
    if (e >= E) return;
    int lane = threadIdx.x & 31;
    size_t rb = (size_t)r * E;
    int s = src[rb + e];
    int d = dst[rb + e];
    float w = ew[rb + e];
    float4 v = ((const float4*)(h + (size_t)s * HID))[lane];
    float* ad = agg + ((size_t)r * NN + d) * HID + lane * 4;
    atomicAdd(ad + 0, v.x * w);
    atomicAdd(ad + 1, v.y * w);
    atomicAdd(ad + 2, v.z * w);
    atomicAdd(ad + 3, v.w * w);
    if (lane == 0) atomicAdd(deg + (size_t)r * NN + d, 1.0f);
}

// agg / max(deg,1) -> f16  (over R*N*HID elements)
__global__ void norm_to_f16(const float* __restrict__ agg, const float* __restrict__ deg,
                            _Float16* __restrict__ out, size_t total)
{
    size_t i = (size_t)blockIdx.x * blockDim.x + threadIdx.x;
    if (i >= total) return;
    size_t rn = i >> 7; // (r*N + n)
    float d = fmaxf(deg[rn], 1.0f);
    out[i] = (_Float16)(agg[i] / d);
}

// ---------------------------------------------------------------------------
// Relation attention + residual + LayerNorm (one block of 128 per node)
// ---------------------------------------------------------------------------
__global__ void __launch_bounds__(128)
rel_attn_ln(const float* __restrict__ stack, const float* __restrict__ q,
            const float* __restrict__ hin,
            const float* __restrict__ g, const float* __restrict__ b,
            float* __restrict__ houtf, _Float16* __restrict__ houth)
{
    __shared__ float red[128];
    __shared__ float sc[3];
    __shared__ float mv[2];
    int n = blockIdx.x, k = threadIdx.x;
    size_t stride = (size_t)NN * HID;
    size_t base = (size_t)n * HID + k;
    float s0 = stack[base];
    float s1 = stack[stride + base];
    float s2 = stack[2 * stride + base];
    float qk = q[k];
    float sv[3] = {s0, s1, s2};
    for (int r = 0; r < 3; ++r) {
        red[k] = tanhf(sv[r]) * qk;
        __syncthreads();
        for (int off = 64; off > 0; off >>= 1) {
            if (k < off) red[k] += red[k + off];
            __syncthreads();
        }
        if (k == 0) sc[r] = red[0];
        __syncthreads();
    }
    float m = fmaxf(sc[0], fmaxf(sc[1], sc[2]));
    float e0 = expf(sc[0] - m), e1 = expf(sc[1] - m), e2 = expf(sc[2] - m);
    float inv = 1.0f / (e0 + e1 + e2);
    float attn = (e0 * s0 + e1 * s1 + e2 * s2) * inv;
    float x = hin[base] + fmaxf(attn, 0.0f);

    red[k] = x; __syncthreads();
    for (int off = 64; off > 0; off >>= 1) {
        if (k < off) red[k] += red[k + off];
        __syncthreads();
    }
    if (k == 0) mv[0] = red[0] * (1.0f / 128.0f);
    __syncthreads();
    float dx = x - mv[0];
    red[k] = dx * dx; __syncthreads();
    for (int off = 64; off > 0; off >>= 1) {
        if (k < off) red[k] += red[k + off];
        __syncthreads();
    }
    if (k == 0) mv[1] = red[0] * (1.0f / 128.0f);
    __syncthreads();
    float y = dx * rsqrtf(mv[1] + 1e-5f) * g[k] + b[k];
    if (houtf) houtf[base] = y;
    if (houth) houth[base] = (_Float16)y;
}

// ---------------------------------------------------------------------------
// GRU element-wise step. gi/gh are [N,384]; hprev updated in place.
// ---------------------------------------------------------------------------
__global__ void gru_elem(const float* __restrict__ gi, const float* __restrict__ gh,
                         float* __restrict__ hprev, float* __restrict__ outs_t,
                         _Float16* __restrict__ hcur16, size_t total)
{
    size_t i = (size_t)blockIdx.x * blockDim.x + threadIdx.x;
    if (i >= total) return;
    size_t n = i >> 7;
    int k = (int)(i & 127);
    size_t b3 = n * 384 + k;
    float ir = gi[b3], iz = gi[b3 + 128], in = gi[b3 + 256];
    float hr = gh[b3], hz = gh[b3 + 128], hn = gh[b3 + 256];
    float r  = 1.0f / (1.0f + expf(-(ir + hr)));
    float zg = 1.0f / (1.0f + expf(-(iz + hz)));
    float nn = tanhf(in + r * hn);
    float h  = hprev[i];
    float hnew = (1.0f - zg) * nn + zg * h;
    hprev[i]  = hnew;
    outs_t[i] = hnew;
    hcur16[i] = (_Float16)hnew;
}

// ---------------------------------------------------------------------------
// Temporal attention over T=4 + concat z = [last, attn, tab] (f16, [N,384])
// ---------------------------------------------------------------------------
__global__ void __launch_bounds__(128)
temporal_concat(const float* __restrict__ outs, const float* __restrict__ tq,
                const float* __restrict__ tab, _Float16* __restrict__ z16)
{
    __shared__ float red[128];
    __shared__ float sc[4];
    int n = blockIdx.x, k = threadIdx.x;
    size_t stride = (size_t)NN * HID;
    size_t base = (size_t)n * HID + k;
    float o[4];
    o[0] = outs[base];
    o[1] = outs[stride + base];
    o[2] = outs[2 * stride + base];
    o[3] = outs[3 * stride + base];
    float qk = tq[k];
    for (int t = 0; t < 4; ++t) {
        red[k] = tanhf(o[t]) * qk;
        __syncthreads();
        for (int off = 64; off > 0; off >>= 1) {
            if (k < off) red[k] += red[k + off];
            __syncthreads();
        }
        if (k == 0) sc[t] = red[0];
        __syncthreads();
    }
    float m = fmaxf(fmaxf(sc[0], sc[1]), fmaxf(sc[2], sc[3]));
    float e0 = expf(sc[0] - m), e1 = expf(sc[1] - m);
    float e2 = expf(sc[2] - m), e3 = expf(sc[3] - m);
    float inv = 1.0f / (e0 + e1 + e2 + e3);
    float attn = (e0 * o[0] + e1 * o[1] + e2 * o[2] + e3 * o[3]) * inv;
    _Float16* zr = z16 + (size_t)n * 384;
    zr[k]       = (_Float16)o[3];       // last
    zr[128 + k] = (_Float16)attn;       // temporal attention
    zr[256 + k] = (_Float16)tab[base];  // tab MLP
}

// ---------------------------------------------------------------------------
// Final projection: out[n] = dot(zh[n,:], cW2) + cb2
// ---------------------------------------------------------------------------
__global__ void __launch_bounds__(128)
final_dot(const float* __restrict__ zh, const float* __restrict__ w,
          const float* __restrict__ b, float* __restrict__ out)
{
    __shared__ float red[128];
    int n = blockIdx.x, k = threadIdx.x;
    red[k] = zh[(size_t)n * HID + k] * w[k];
    __syncthreads();
    for (int off = 64; off > 0; off >>= 1) {
        if (k < off) red[k] += red[k + off];
        __syncthreads();
    }
    if (k == 0) out[n] = red[0] + b[0];
}

// ---------------------------------------------------------------------------
// Host-side helpers
// ---------------------------------------------------------------------------
static inline void launch_gemm(hipStream_t st, const _Float16* A, int lda, int ktiles,
                               const _Float16* Bpk, const float* bias,
                               float* Cf, _Float16* Ch, int ldc, int kout, int relu)
{
    dim3 grid(NN / 16, kout / 128, 1); // 8 N-tiles per block (8 waves)
    wmma_gemm<<<grid, 256, 0, st>>>(A, lda, ktiles, Bpk, bias, Cf, Ch, ldc, relu);
}

static inline void launch_pack(hipStream_t st, const float* W, _Float16* dstp, int K, int Kout)
{
    int total = K * Kout;
    pack_b<<<(total + 255) / 256, 256, 0, st>>>(W, dstp, K, Kout, total);
}

static inline void launch_zero(hipStream_t st, float* p, size_t n)
{
    zero_f32<<<(unsigned)((n + 255) / 256), 256, 0, st>>>(p, n);
}

extern "C" void kernel_launch(void* const* d_in, const int* in_sizes, int n_in,
                              void* d_out, int out_size, void* d_ws, size_t ws_size,
                              hipStream_t stream)
{
    (void)in_sizes; (void)n_in; (void)out_size; (void)ws_size;

    const float* feat  = (const float*)d_in[0];
    const int*   src   = (const int*)d_in[1];
    const int*   dstp  = (const int*)d_in[2];
    const float* ew    = (const float*)d_in[3];
    const float* W_in  = (const float*)d_in[4];
    const float* b_in  = (const float*)d_in[5];
    const float* W1    = (const float*)d_in[6];
    const float* b1    = (const float*)d_in[7];
    const float* W2    = (const float*)d_in[8];
    const float* b2    = (const float*)d_in[9];
    const float* q1    = (const float*)d_in[10];
    const float* q2    = (const float*)d_in[11];
    const float* ln1g  = (const float*)d_in[12];
    const float* ln1b  = (const float*)d_in[13];
    const float* ln2g  = (const float*)d_in[14];
    const float* ln2b  = (const float*)d_in[15];
    const float* gWih  = (const float*)d_in[16];
    const float* gbih  = (const float*)d_in[17];
    const float* gWhh  = (const float*)d_in[18];
    const float* gbhh  = (const float*)d_in[19];
    const float* tq    = (const float*)d_in[20];
    const float* tW1   = (const float*)d_in[21];
    const float* tb1   = (const float*)d_in[22];
    const float* tW2   = (const float*)d_in[23];
    const float* tb2   = (const float*)d_in[24];
    const float* cW1   = (const float*)d_in[25];
    const float* cb1   = (const float*)d_in[26];
    const float* cW2   = (const float*)d_in[27];
    const float* cb2   = (const float*)d_in[28];
    float* out = (float*)d_out;

    // ---- workspace layout (bump allocator, 256B aligned) ----
    const size_t NH = (size_t)NN * HID;        // 6.4M elems
    char* base = (char*)d_ws;
    size_t woff = 0;
    auto alloc = [&](size_t bytes) -> char* {
        char* p = base + woff;
        woff = (woff + bytes + 255) & ~(size_t)255;
        return p;
    };
    _Float16* feat16 = (_Float16*)alloc(TT * NH * sizeof(_Float16));  // f16 features
    _Float16* hs16   = (_Float16*)alloc(TT * NH * sizeof(_Float16));  // spatial outputs (GRU input)
    float*    outs   = (float*)   alloc(TT * NH * sizeof(float));     // GRU outputs
    float*    bigA   = (float*)   alloc(RR * NH * sizeof(float));     // agg   | gi [N,384]
    float*    bigB   = (float*)   alloc(RR * NH * sizeof(float));     // stack | gh [N,384]
    _Float16* h16X   = (_Float16*)alloc(RR * NH * sizeof(_Float16));  // aggn16 | z16 [N,384]
    float*    hA     = (float*)   alloc(NH * sizeof(float));          // h0 | tab_f32
    float*    hB     = (float*)   alloc(NH * sizeof(float));          // h1 | hprev | zh
    float*    deg    = (float*)   alloc((size_t)RR * NN * sizeof(float));
    _Float16* h16Y   = (_Float16*)alloc(NH * sizeof(_Float16));       // tmp16 | hcur16
    _Float16* pk     = (_Float16*)alloc((size_t)294912 * sizeof(_Float16));

    _Float16* pk_Win  = pk;
    _Float16* pk_W1   = pk + 16384;
    _Float16* pk_W2   = pk + 65536;
    _Float16* pk_gWih = pk + 114688;
    _Float16* pk_gWhh = pk + 163840;
    _Float16* pk_tW1  = pk + 212992;
    _Float16* pk_tW2  = pk + 229376;
    _Float16* pk_cW1  = pk + 245760;

    // ---- pack all weights into WMMA B-fragment layout (f16) ----
    launch_pack(stream, W_in, pk_Win, 128, 128);
    for (int r = 0; r < RR; ++r) {
        launch_pack(stream, W1 + (size_t)r * 16384, pk_W1 + (size_t)r * 16384, 128, 128);
        launch_pack(stream, W2 + (size_t)r * 16384, pk_W2 + (size_t)r * 16384, 128, 128);
    }
    launch_pack(stream, gWih, pk_gWih, 128, 384);
    launch_pack(stream, gWhh, pk_gWhh, 128, 384);
    launch_pack(stream, tW1,  pk_tW1,  128, 128);
    launch_pack(stream, tW2,  pk_tW2,  128, 128);
    launch_pack(stream, cW1,  pk_cW1,  384, 128);

    // ---- features -> f16 once ----
    {
        size_t tot = TT * NH;
        f32_to_f16<<<(unsigned)((tot + 255) / 256), 256, 0, stream>>>(feat, feat16, tot);
    }

    dim3 egrid((EE + 7) / 8, RR, 1);

    // ---- spatial stage (per timestep) ----
    for (int t = 0; t < TT; ++t) {
        // h0 = relu(feat @ W_in + b_in)
        launch_gemm(stream, feat16 + (size_t)t * NH, 128, 4, pk_Win, b_in,
                    hA, nullptr, 128, 128, 1);

        const int*   st = src  + (size_t)t * RR * EE;
        const int*   dt = dstp + (size_t)t * RR * EE;
        const float* wt = ew   + (size_t)t * RR * EE;

        // ---- layer 1 ----
        launch_zero(stream, bigA, RR * NH);
        launch_zero(stream, deg, (size_t)RR * NN);
        edge_scatter<<<egrid, 256, 0, stream>>>(hA, st, dt, wt, bigA, deg, EE);
        norm_to_f16<<<(unsigned)((RR * NH + 255) / 256), 256, 0, stream>>>(bigA, deg, h16X, RR * NH);
        for (int r = 0; r < RR; ++r)
            launch_gemm(stream, h16X + (size_t)r * NH, 128, 4,
                        pk_W1 + (size_t)r * 16384, b1 + (size_t)r * 128,
                        bigB + (size_t)r * NH, nullptr, 128, 128, 0);
        rel_attn_ln<<<NN, 128, 0, stream>>>(bigB, q1, hA, ln1g, ln1b, hB, nullptr);

        // ---- layer 2 ----
        launch_zero(stream, bigA, RR * NH);
        launch_zero(stream, deg, (size_t)RR * NN);
        edge_scatter<<<egrid, 256, 0, stream>>>(hB, st, dt, wt, bigA, deg, EE);
        norm_to_f16<<<(unsigned)((RR * NH + 255) / 256), 256, 0, stream>>>(bigA, deg, h16X, RR * NH);
        for (int r = 0; r < RR; ++r)
            launch_gemm(stream, h16X + (size_t)r * NH, 128, 4,
                        pk_W2 + (size_t)r * 16384, b2 + (size_t)r * 128,
                        bigB + (size_t)r * NH, nullptr, 128, 128, 0);
        rel_attn_ln<<<NN, 128, 0, stream>>>(bigB, q2, hB, ln2g, ln2b,
                                            nullptr, hs16 + (size_t)t * NH);
    }

    // ---- tab MLP: relu(relu(feat[T-1] @ tW1 + tb1) @ tW2 + tb2) -> hA ----
    launch_gemm(stream, feat16 + (size_t)(TT - 1) * NH, 128, 4, pk_tW1, tb1,
                nullptr, h16Y, 128, 128, 1);
    launch_gemm(stream, h16Y, 128, 4, pk_tW2, tb2, hA, nullptr, 128, 128, 1);

    // ---- GRU over T=4 ----
    launch_zero(stream, hB, NH);                       // hprev (f32) = 0
    launch_zero(stream, (float*)h16Y, NH / 2);         // hcur16 = 0
    for (int t = 0; t < TT; ++t) {
        launch_gemm(stream, hs16 + (size_t)t * NH, 128, 4, pk_gWih, gbih,
                    bigA, nullptr, 384, 384, 0);       // gi
        launch_gemm(stream, h16Y, 128, 4, pk_gWhh, gbhh,
                    bigB, nullptr, 384, 384, 0);       // gh
        gru_elem<<<(unsigned)((NH + 255) / 256), 256, 0, stream>>>(
            bigA, bigB, hB, outs + (size_t)t * NH, h16Y, NH);
    }

    // ---- temporal attention + concat -> z16 [N,384] ----
    temporal_concat<<<NN, 128, 0, stream>>>(outs, tq, hA, h16X);

    // ---- final: zh = relu(z @ cW1 + cb1); out = zh . cW2 + cb2 ----
    launch_gemm(stream, h16X, 384, 12, pk_cW1, cb1, hB, nullptr, 128, 128, 1);
    final_dot<<<NN, 128, 0, stream>>>(hB, cW2, cb2, out);
}
